// EdgeMaskNet_53163105190631
// MI455X (gfx1250) — matrix operand order repaired and checked by reading, not compiled, not run on hardware
//
#include <hip/hip_runtime.h>

#define N_NODES 100000
#define N_EDGES 1600000
#define F_IN    64
#define H       72
#define F_E     8
#define L_LAYERS 3
#define BN_EPS  1e-5f

typedef __attribute__((ext_vector_type(16))) _Float16 v16h;
typedef __attribute__((ext_vector_type(8)))  _Float16 v8h;
typedef __attribute__((ext_vector_type(8)))  float    v8f;
typedef __attribute__((ext_vector_type(4)))  unsigned int v4u;

// ---------------------------------------------------------------- utilities
__global__ void k_zero(float* __restrict__ p, long long n) {
  long long i = (long long)blockIdx.x * blockDim.x + threadIdx.x;
  if (i < n) p[i] = 0.f;
}

__global__ void k_deg(const long long* __restrict__ ei, float* __restrict__ deg) {
  int e = blockIdx.x * blockDim.x + threadIdx.x;
  if (e < N_EDGES) atomicAdd(&deg[ei[(long long)N_EDGES + e]], 1.f);
}

__global__ void k_dinv(float* __restrict__ deg) {
  int i = blockIdx.x * blockDim.x + threadIdx.x;
  if (i < N_NODES) { float d = deg[i]; deg[i] = d > 0.f ? rsqrtf(d) : 0.f; }
}

// ------------------------------------------------- node linear + ARMA layers
__global__ __launch_bounds__(96) void k_node_lin(
    const float* __restrict__ x, const float* __restrict__ w,
    const float* __restrict__ b, float* __restrict__ h) {
  __shared__ float xs[F_IN];
  int i = blockIdx.x, j = threadIdx.x;
  if (j < F_IN) xs[j] = x[(long long)i * F_IN + j];
  __syncthreads();
  if (j < H) {
    float acc = b[j];
#pragma unroll 8
    for (int k = 0; k < F_IN; k++) acc = fmaf(xs[k], w[k * H + j], acc);
    h[(long long)i * H + j] = fmaxf(acc, 0.f);
  }
}

__global__ __launch_bounds__(96) void k_gemm72(
    const float* __restrict__ hin, const float* __restrict__ W,
    float* __restrict__ out) {
  __shared__ float hs[H];
  int i = blockIdx.x, j = threadIdx.x;
  if (j < H) hs[j] = hin[(long long)i * H + j];
  __syncthreads();
  if (j < H) {
    float acc = 0.f;
#pragma unroll 8
    for (int k = 0; k < H; k++) acc = fmaf(hs[k], W[k * H + j], acc);
    out[(long long)i * H + j] = acc;
  }
}

// one warp per edge: agg[dst] += norm * t[src]
__global__ void k_scatter(const long long* __restrict__ ei,
                          const float* __restrict__ dinv,
                          const float* __restrict__ t,
                          float* __restrict__ agg) {
  long long gt = (long long)blockIdx.x * blockDim.x + threadIdx.x;
  long long e = gt >> 5;
  int lane = (int)(gt & 31);
  if (e >= N_EDGES) return;
  long long s = ei[e], d = ei[N_EDGES + e];
  float w = dinv[s] * dinv[d];
  for (int c = lane; c < H; c += 32)
    atomicAdd(&agg[d * H + c], w * t[s * H + c]);
}

__global__ __launch_bounds__(96) void k_update(
    const float* __restrict__ agg, const float* __restrict__ Wr,
    const float* __restrict__ b, float* __restrict__ h) {
  __shared__ float hs[H];
  int i = blockIdx.x, j = threadIdx.x;
  if (j < H) hs[j] = h[(long long)i * H + j];
  __syncthreads();
  if (j < H) {
    float acc = b[j] + agg[(long long)i * H + j];
#pragma unroll 8
    for (int k = 0; k < H; k++) acc = fmaf(hs[k], Wr[k * H + j], acc);
    h[(long long)i * H + j] = fmaxf(acc, 0.f);
  }
}

__global__ __launch_bounds__(288) void k_bnstat(
    const float* __restrict__ h, float* __restrict__ bnsum,
    float* __restrict__ bnsq) {
  __shared__ float ssum[H], ssq[H];
  int tid = threadIdx.x;
  int c = tid % H, r0 = tid / H;  // 0..3
  if (tid < H) { ssum[tid] = 0.f; ssq[tid] = 0.f; }
  __syncthreads();
  float ls = 0.f, lq = 0.f;
  for (long long r = (long long)blockIdx.x * 4 + r0; r < N_NODES;
       r += (long long)gridDim.x * 4) {
    float v = h[r * H + c];
    ls += v; lq += v * v;
  }
  atomicAdd(&ssum[c], ls);
  atomicAdd(&ssq[c], lq);
  __syncthreads();
  if (tid < H) { atomicAdd(&bnsum[tid], ssum[tid]); atomicAdd(&bnsq[tid], ssq[tid]); }
}

__global__ void k_bnfinal(const float* __restrict__ bnsum,
                          const float* __restrict__ bnsq,
                          const float* __restrict__ gamma,
                          const float* __restrict__ beta,
                          float* __restrict__ scale, float* __restrict__ shift) {
  int c = threadIdx.x;
  if (c < H) {
    float m = bnsum[c] * (1.f / N_NODES);
    float v = bnsq[c] * (1.f / N_NODES) - m * m;
    float s = gamma[c] * rsqrtf(v + BN_EPS);
    scale[c] = s;
    shift[c] = beta[c] - m * s;
  }
}

__global__ void k_bnapply(float* __restrict__ h, const float* __restrict__ scale,
                          const float* __restrict__ shift) {
  long long gid = (long long)blockIdx.x * blockDim.x + threadIdx.x;
  if (gid < (long long)N_NODES * H) {
    int c = (int)(gid % H);
    h[gid] = fmaf(h[gid], scale[c], shift[c]);
  }
}

// -------------------------------------------------------- edge-phase prep
// h (f32, N x 72) -> hp16 (f16, N x 80, zero-padded)
__global__ void k_tof16(const float* __restrict__ h, _Float16* __restrict__ hp) {
  long long gid = (long long)blockIdx.x * blockDim.x + threadIdx.x;
  if (gid >= (long long)N_NODES * 80) return;
  long long r = gid / 80;
  int c = (int)(gid % 80);
  hp[gid] = (c < H) ? (_Float16)h[r * H + c] : (_Float16)0.f;
}

// Pack e1_w / mlp_w1 ([144][72] f32) into WMMA B-fragment order (f16):
// layout [m][nt][ks][lane][16]; lane<16 holds K 0..15 of k-step, lane>=16 K 16..31,
// col = nt*16 + lane%16.  A-column a maps to weight row: a<72 -> a ; 80<=a<152 -> a-8.
__global__ void k_packw(const float* __restrict__ e1w,
                        const float* __restrict__ mlpw1,
                        _Float16* __restrict__ packW) {
  int gid = blockIdx.x * blockDim.x + threadIdx.x;
  if (gid >= 2 * 5 * 5 * 32 * 16) return;
  int idx  = gid & 15;
  int lane = (gid >> 4) & 31;
  int ks   = (gid >> 9) % 5;
  int nt   = ((gid >> 9) / 5) % 5;
  int m    = gid / (5 * 5 * 32 * 16);
  int a    = ks * 32 + ((lane >> 4) << 4) + idx;   // A column 0..159
  int col  = nt * 16 + (lane & 15);
  int wrow = (a < H) ? a : ((a >= 80 && a < 80 + H) ? (a - 8) : -1);
  const float* W = m ? mlpw1 : e1w;
  float v = (wrow >= 0 && col < H) ? W[wrow * H + col] : 0.f;
  packW[gid] = (_Float16)v;
}

// ------------------------------------------------------------ fused edge MLP
// one wave = 16 edges; A rows staged in LDS as [src(40)|unwritten(40)|dst(40)|unwritten(40)] f16,
// GEMM1 (e1) -> stage [e1(72)|0|e2(72)|unwritten] -> GEMM2 (mlp_w1) -> tanh -> dot w2.
__global__ __launch_bounds__(256) void k_edge(
    const _Float16* __restrict__ hp16, const long long* __restrict__ ei,
    const float* __restrict__ edge_attr, const _Float16* __restrict__ packW,
    const float* __restrict__ e1_b, const float* __restrict__ e2_w,
    const float* __restrict__ e2_b, const float* __restrict__ mlp_b1,
    const float* __restrict__ mlp_w2, const float* __restrict__ mlp_b2,
    float* __restrict__ out) {
  __shared__ __attribute__((aligned(16))) _Float16 stage[8][16][160];
  __shared__ float s_e2w[F_E * H];
  __shared__ float s_e1b[H], s_e2b[H], s_b1[H], s_w2[H];
  __shared__ float s_b2;

  int tid = threadIdx.x;
  for (int i = tid; i < F_E * H; i += 256) s_e2w[i] = e2_w[i];
  if (tid < H) {
    s_e1b[tid] = e1_b[tid]; s_e2b[tid] = e2_b[tid];
    s_b1[tid]  = mlp_b1[tid]; s_w2[tid] = mlp_w2[tid];
  }
  if (tid == 0) s_b2 = mlp_b2[0];

  int wave = tid >> 5, lane = tid & 31;
  int er  = lane & 15;   // edge row within tile
  int grp = lane >> 4;   // 0 / 1
  long long ebase = ((long long)blockIdx.x * 8 + wave) * 16;
  _Float16(*st)[160] = stage[wave];

  // gather first 40 f16: grp0 -> h[src] into cols 0..39, grp1 -> h[dst] into cols 80..119
  {
    long long nid = ei[(long long)grp * N_EDGES + ebase + er];
    const v4u* sp = (const v4u*)(hp16 + nid * 80);
    v4u* dp = (v4u*)(&st[er][grp * 80]);
#pragma unroll
    for (int q = 0; q < 5; q++) dp[q] = sp[q];
  }
  __syncthreads();

  const _Float16* rowp = &st[er][0];
  v8f C[5];
#pragma unroll
  for (int nt = 0; nt < 5; nt++)
#pragma unroll
    for (int q = 0; q < 8; q++) C[nt][q] = 0.f;

  // GEMM1: [16x160] x e1_w -> C (5 n-tiles of 16)
#pragma unroll
  for (int ks = 0; ks < 5; ks++) {
    int o1 = ks * 32 + grp * 16;  // A-frag: lane<16 K 0..15, lane>=16 K 16..31
    v8h lo = *(const v8h*)(rowp + o1);
    v8h hi = *(const v8h*)(rowp + o1 + 8);
    v16h a = __builtin_shufflevector(lo, hi, 0, 1, 2, 3, 4, 5, 6, 7, 8, 9, 10,
                                     11, 12, 13, 14, 15);
#pragma unroll
    for (int nt = 0; nt < 5; nt++) {
      v16h b = *(const v16h*)(packW + (((0 * 5 + nt) * 5 + ks) * 32 + lane) * 16);
      C[nt] = __builtin_amdgcn_wmma_f32_16x16x32_f16(false, a, false, b,
                                                     (short)0, C[nt], false, false);
    }
  }
  __syncthreads();

  // write e1 + bias back to stage cols 0..79 as f16
#pragma unroll
  for (int nt = 0; nt < 5; nt++) {
    int col = nt * 16 + er;
    float bc = (col < H) ? s_e1b[col] : 0.f;
#pragma unroll
    for (int i = 0; i < 8; i++) {
      int row = grp ? (i + 8) : i;
      float v = (col < H) ? (C[nt][i] + bc) : 0.f;
      st[row][col] = (_Float16)v;
    }
  }
  // e2 = edge_attr @ e2_w + b into cols 80..151 (cols 152..159 are never written)
  {
    long long eg = ebase + er;
    float ea[F_E];
#pragma unroll
    for (int k = 0; k < F_E; k++) ea[k] = edge_attr[eg * F_E + k];
    for (int c = grp * 36; c < grp * 36 + 36; c++) {
      float acc = s_e2b[c];
#pragma unroll
      for (int k = 0; k < F_E; k++) acc = fmaf(ea[k], s_e2w[k * H + c], acc);
      st[er][80 + c] = (_Float16)acc;
    }
  }
  __syncthreads();

  // GEMM2: [e1|e2] x mlp_w1
#pragma unroll
  for (int nt = 0; nt < 5; nt++)
#pragma unroll
    for (int q = 0; q < 8; q++) C[nt][q] = 0.f;
#pragma unroll
  for (int ks = 0; ks < 5; ks++) {
    int o1 = ks * 32 + grp * 16;
    v8h lo = *(const v8h*)(rowp + o1);
    v8h hi = *(const v8h*)(rowp + o1 + 8);
    v16h a = __builtin_shufflevector(lo, hi, 0, 1, 2, 3, 4, 5, 6, 7, 8, 9, 10,
                                     11, 12, 13, 14, 15);
#pragma unroll
    for (int nt = 0; nt < 5; nt++) {
      v16h b = *(const v16h*)(packW + (((1 * 5 + nt) * 5 + ks) * 32 + lane) * 16);
      C[nt] = __builtin_amdgcn_wmma_f32_16x16x32_f16(false, a, false, b,
                                                     (short)0, C[nt], false, false);
    }
  }

  // out = tanh(C + b1) . w2 + b2 ; C layout: VGPR i -> row i (lanes<16) / i+8, col = nt*16+lane%16
  float partial[8];
#pragma unroll
  for (int i = 0; i < 8; i++) partial[i] = 0.f;
#pragma unroll
  for (int nt = 0; nt < 5; nt++) {
    int col = nt * 16 + er;
    if (col < H) {
      float wc = s_w2[col], bc = s_b1[col];
#pragma unroll
      for (int i = 0; i < 8; i++)
        partial[i] = fmaf(tanhf(C[nt][i] + bc), wc, partial[i]);
    }
  }
#pragma unroll
  for (int i = 0; i < 8; i++) {
    float p = partial[i];
    p += __shfl_xor(p, 1, 32);
    p += __shfl_xor(p, 2, 32);
    p += __shfl_xor(p, 4, 32);
    p += __shfl_xor(p, 8, 32);
    partial[i] = p;
  }
  if (er == 0) {
    int rbase = grp * 8;
#pragma unroll
    for (int i = 0; i < 8; i++) out[ebase + rbase + i] = partial[i] + s_b2;
  }
}

// ------------------------------------------------------------------ launcher
extern "C" void kernel_launch(void* const* d_in, const int* in_sizes, int n_in,
                              void* d_out, int out_size, void* d_ws, size_t ws_size,
                              hipStream_t stream) {
  (void)in_sizes; (void)n_in; (void)out_size; (void)ws_size;
  const float*      x         = (const float*)d_in[0];
  const long long*  ei        = (const long long*)d_in[1];
  const float*      edge_attr = (const float*)d_in[2];
  const float*      node_w    = (const float*)d_in[3];
  const float*      node_b    = (const float*)d_in[4];
  const float*      arma_iw   = (const float*)d_in[5];
  const float*      arma_rw   = (const float*)d_in[6];
  const float*      arma_b    = (const float*)d_in[7];
  const float*      bn_gamma  = (const float*)d_in[8];
  const float*      bn_beta   = (const float*)d_in[9];
  const float*      e1_w      = (const float*)d_in[10];
  const float*      e1_b      = (const float*)d_in[11];
  const float*      e2_w      = (const float*)d_in[12];
  const float*      e2_b      = (const float*)d_in[13];
  const float*      mlp_w1    = (const float*)d_in[14];
  const float*      mlp_b1    = (const float*)d_in[15];
  const float*      mlp_w2    = (const float*)d_in[16];
  const float*      mlp_b2    = (const float*)d_in[17];
  float* out = (float*)d_out;

  float* h       = (float*)d_ws;
  float* t       = h + (size_t)N_NODES * H;
  float* agg     = t + (size_t)N_NODES * H;
  float* deg     = agg + (size_t)N_NODES * H;
  float* bnsum   = deg + N_NODES;
  float* bnsq    = bnsum + H;
  float* bnscale = bnsq + H;
  float* bnshift = bnscale + H;
  _Float16* hp16  = (_Float16*)t;    // overlays t after node phase
  _Float16* packW = (_Float16*)agg;  // overlays agg after node phase

  k_zero<<<(N_NODES + 255) / 256, 256, 0, stream>>>(deg, (long long)N_NODES);
  k_deg<<<N_EDGES / 256, 256, 0, stream>>>(ei, deg);
  k_dinv<<<(N_NODES + 255) / 256, 256, 0, stream>>>(deg);
  k_node_lin<<<N_NODES, 96, 0, stream>>>(x, node_w, node_b, h);

  for (int l = 0; l < L_LAYERS; l++) {
    k_gemm72<<<N_NODES, 96, 0, stream>>>(h, arma_iw + (size_t)l * H * H, t);
    k_zero<<<((long long)N_NODES * H) / 256, 256, 0, stream>>>(agg, (long long)N_NODES * H);
    k_scatter<<<((long long)N_EDGES * 32) / 256, 256, 0, stream>>>(ei, deg, t, agg);
    k_update<<<N_NODES, 96, 0, stream>>>(agg, arma_rw + (size_t)l * H * H,
                                         arma_b + (size_t)l * H, h);
    k_zero<<<1, 256, 0, stream>>>(bnsum, (long long)(2 * H));
    k_bnstat<<<512, 288, 0, stream>>>(h, bnsum, bnsq);
    k_bnfinal<<<1, 96, 0, stream>>>(bnsum, bnsq, bn_gamma + (size_t)l * H,
                                    bn_beta + (size_t)l * H, bnscale, bnshift);
    k_bnapply<<<((long long)N_NODES * H) / 256, 256, 0, stream>>>(h, bnscale, bnshift);
  }

  k_tof16<<<((long long)N_NODES * 80) / 256, 256, 0, stream>>>(h, hp16);
  k_packw<<<100, 256, 0, stream>>>(e1_w, mlp_w1, packW);
  k_edge<<<N_EDGES / 128, 256, 0, stream>>>(hp16, ei, edge_attr, packW, e1_b,
                                            e2_w, e2_b, mlp_b1, mlp_w2, mlp_b2, out);
}
